// MGN_1675037245681
// MI455X (gfx1250) — matrix operand order repaired and checked by reading, not compiled, not run on hardware
//
#include <hip/hip_runtime.h>
#include <hip/hip_bf16.h>

#define HD __device__ __forceinline__

typedef __bf16 bf16;
typedef __attribute__((ext_vector_type(8)))  __bf16 v8bf;
typedef __attribute__((ext_vector_type(16))) __bf16 v16bf;
typedef __attribute__((ext_vector_type(8)))  float  v8f;

union V16 { v16bf v; v8bf h[2]; };

// A-fragment (16x32 bf16): lane<16 holds K {0..7,16..23}, lane>=16 holds {8..15,24..31}
HD v16bf load_afrag(const bf16* rowp, int kc, int lane) {
  int kb = kc * 32 + ((lane & 16) ? 8 : 0);
  V16 u;
  u.h[0] = *(const v8bf*)(rowp + kb);
  u.h[1] = *(const v8bf*)(rowp + kb + 16);
  return u.v;
}

// B-fragment (32x16 bf16), weights stored N-major [n][kpad]:
// lane<16 -> K {0..7,16..23} of column n=lane&15; lane>=16 -> K {8..15,24..31}
HD v16bf load_bfrag(const bf16* wT, int kpad, int n, int kc, int lane) {
  int kb = kc * 32 + ((lane & 16) ? 8 : 0);
  const bf16* p = wT + n * kpad + kb;
  V16 u;
  u.h[0] = *(const v8bf*)(p);
  u.h[1] = *(const v8bf*)(p + 16);
  return u.v;
}

HD float elu1(float x) { return x > 0.f ? x : __expf(x) - 1.f; }

HD void copy_w(const bf16* __restrict__ g, bf16* __restrict__ s, int elems, int tid) {
  for (int i = tid * 8; i < elems; i += 256 * 8)
    *(v8bf*)(s + i) = *(const v8bf*)(g + i);
}
HD void copy_bias(const float* __restrict__ g, float* __restrict__ s, int n, int tid) {
  if (tid < 128) s[tid] = (tid < n) ? g[tid] : 0.f;
}

// One 128x128 GEMM layer: Aout = act(Ain[128 x 32*KC] x W^T + bias), bf16 in/out, f32 accum
template <int KC>
HD void mlp_layer(const bf16* __restrict__ Ain, int astride,
                  const bf16* __restrict__ wT, const float* __restrict__ bias,
                  bool act, bf16* __restrict__ Aout, int ostride,
                  int lane, int wave) {
  v16bf af[KC];
  const bf16* rowp = Ain + (wave * 16 + (lane & 15)) * astride;
#pragma unroll
  for (int kc = 0; kc < KC; ++kc) af[kc] = load_afrag(rowp, kc, lane);
  int col0 = lane & 15;
  int rofs = wave * 16 + ((lane & 16) ? 8 : 0);
  for (int nt = 0; nt < 8; ++nt) {
    v8f acc = {};
#pragma unroll
    for (int kc = 0; kc < KC; ++kc) {
      v16bf b = load_bfrag(wT, KC * 32, nt * 16 + col0, kc, lane);
      acc = __builtin_amdgcn_wmma_f32_16x16x32_bf16(false, af[kc], false, b,
                                                    (short)0, acc, false, false);
    }
    float bs = bias[nt * 16 + col0];
#pragma unroll
    for (int v = 0; v < 8; ++v) {
      float y = acc[v] + bs;
      if (act) y = elu1(y);
      Aout[(rofs + v) * ostride + nt * 16 + col0] = (bf16)y;
    }
  }
}

// ---------------- Edge MLP fused with scatter-add (persistent blocks) ----------------
// Edge feature layout (W0 rows permuted to match in prep):
//   cols 0..7  = x[src]   (async b128 -> LDS, 16B aligned)
//   cols 8..15 = x[dst]   (async b128 -> LDS, 16B aligned)
//   cols 16..18 = pos[dst]-pos[src]
//   cols 19..31 = 0
__global__ __launch_bounds__(256)
void edge_kernel(const bf16* __restrict__ xb, const float* __restrict__ pos,
                 const int* __restrict__ eidx, int E, int ntiles,
                 const bf16* __restrict__ W0T, const float* __restrict__ b0,
                 const bf16* __restrict__ W2T, const float* __restrict__ b2,
                 const bf16* __restrict__ W3T, const float* __restrict__ b3,
                 float* __restrict__ es) {
  __shared__ bf16 sIn[128 * 32];
  __shared__ bf16 sH1[128 * 128];
  __shared__ bf16 sH2[128 * 128];
  __shared__ bf16 sW0[128 * 32];
  __shared__ bf16 sW2[128 * 128];
  __shared__ bf16 sW3[128 * 128];
  __shared__ float sB0[128], sB2[128], sB3[128];
  __shared__ int sSrc[128];

  int tid = threadIdx.x, lane = tid & 31, wave = tid >> 5;

  // Stage all edge-MLP weights once per persistent block.
  copy_w(W0T, sW0, 128 * 32, tid);
  copy_w(W2T, sW2, 128 * 128, tid);
  copy_w(W3T, sW3, 128 * 128, tid);
  copy_bias(b0, sB0, 128, tid);
  copy_bias(b2, sB2, 128, tid);
  copy_bias(b3, sB3, 128, tid);
  if (tid < 128) {  // constant zero padding columns, written once
    bf16* row = sIn + tid * 32;
#pragma unroll
    for (int i = 19; i < 32; ++i) row[i] = (bf16)0.f;
  }
  __syncthreads();

  for (int tile = blockIdx.x; tile < ntiles; tile += gridDim.x) {
    // Prefetch next tile's indices while we work on this one.
    if (tid < 128) {
      long pe = (long)(tile + gridDim.x) * 128 + tid;
      if (pe < E) {
        __builtin_prefetch(eidx + pe, 0, 3);
        __builtin_prefetch(eidx + (long)E + pe, 0, 3);
      }
    }
    // Build the input tile: async gather of x rows straight into LDS.
    if (tid < 128) {
      long e = (long)tile * 128 + tid;
      bf16* row = sIn + tid * 32;
      if (e < E) {
        int s = eidx[e], d = eidx[(long)E + e];
        sSrc[tid] = s;
        unsigned lds0 = (unsigned)(uintptr_t)row;
        unsigned long long gas = (unsigned long long)(uintptr_t)(xb + (long)s * 8);
        unsigned long long gad = (unsigned long long)(uintptr_t)(xb + (long)d * 8);
        asm volatile("global_load_async_to_lds_b128 %0, %1, off"
                     :: "v"(lds0), "v"(gas) : "memory");
        asm volatile("global_load_async_to_lds_b128 %0, %1, off"
                     :: "v"(lds0 + 16u), "v"(gad) : "memory");
        row[16] = (bf16)(pos[d * 3 + 0] - pos[s * 3 + 0]);
        row[17] = (bf16)(pos[d * 3 + 1] - pos[s * 3 + 1]);
        row[18] = (bf16)(pos[d * 3 + 2] - pos[s * 3 + 2]);
      } else {
        sSrc[tid] = -1;
        for (int i = 0; i < 19; ++i) row[i] = (bf16)0.f;
      }
    }
    asm volatile("s_wait_asynccnt 0x0" ::: "memory");
    __syncthreads();

    mlp_layer<1>(sIn, 32, sW0, sB0, true, sH1, 128, lane, wave);
    __syncthreads();
    mlp_layer<4>(sH1, 128, sW2, sB2, true, sH2, 128, lane, wave);
    __syncthreads();

    // Final layer: compute edge_attr tile and scatter-add to edge_sum[src]
    {
      v16bf af[4];
      const bf16* rowp = sH2 + (wave * 16 + (lane & 15)) * 128;
#pragma unroll
      for (int kc = 0; kc < 4; ++kc) af[kc] = load_afrag(rowp, kc, lane);
      int col0 = lane & 15;
      int rofs = wave * 16 + ((lane & 16) ? 8 : 0);
      for (int nt = 0; nt < 8; ++nt) {
        v8f acc = {};
#pragma unroll
        for (int kc = 0; kc < 4; ++kc) {
          v16bf b = load_bfrag(sW3, 128, nt * 16 + col0, kc, lane);
          acc = __builtin_amdgcn_wmma_f32_16x16x32_bf16(false, af[kc], false, b,
                                                        (short)0, acc, false, false);
        }
        float bs = sB3[nt * 16 + col0];
#pragma unroll
        for (int v = 0; v < 8; ++v) {
          int s = sSrc[rofs + v];
          if (s >= 0)
            unsafeAtomicAdd(es + (long)s * 128 + nt * 16 + col0, acc[v] + bs);
        }
      }
    }
    __syncthreads();  // protect sIn/sSrc before next tile's build
  }
}

// ---------------- Node MLP + decoder MLP fused ----------------
__global__ __launch_bounds__(256)
void node_kernel(const bf16* __restrict__ xb, const float* __restrict__ es, int N,
                 const bf16* nW0T, const float* nb0,
                 const bf16* nW2T, const float* nb2,
                 const bf16* nW3T, const float* nb3,
                 const bf16* dW0T, const float* db0,
                 const bf16* dW2T, const float* db2,
                 const bf16* dW3T, const float* db3,
                 float* __restrict__ out) {
  __shared__ bf16 sA[128 * 160];
  __shared__ bf16 sB[128 * 128];
  __shared__ bf16 sW[128 * 160];
  __shared__ float sBias[128];

  int tid = threadIdx.x, lane = tid & 31, wave = tid >> 5;
  long g0 = (long)blockIdx.x * 128;

  if (tid < 128) {
    long g = g0 + tid;
    bf16* row = sA + tid * 160;
    if (g < N) {
      v8bf xr = *(const v8bf*)(xb + g * 8);
#pragma unroll
      for (int i = 0; i < 8; ++i) row[i] = xr[i];
      const float* ep = es + g * 128;
#pragma unroll 4
      for (int k = 0; k < 128; k += 4) {
        float4 f = *(const float4*)(ep + k);
        row[8 + k + 0] = (bf16)f.x; row[8 + k + 1] = (bf16)f.y;
        row[8 + k + 2] = (bf16)f.z; row[8 + k + 3] = (bf16)f.w;
      }
#pragma unroll
      for (int i = 136; i < 160; ++i) row[i] = (bf16)0.f;
    } else {
      for (int i = 0; i < 160; ++i) row[i] = (bf16)0.f;
    }
  }
  copy_w(nW0T, sW, 128 * 160, tid); copy_bias(nb0, sBias, 128, tid); __syncthreads();
  mlp_layer<5>(sA, 160, sW, sBias, true, sB, 128, lane, wave); __syncthreads();
  copy_w(nW2T, sW, 128 * 128, tid); copy_bias(nb2, sBias, 128, tid); __syncthreads();
  mlp_layer<4>(sB, 128, sW, sBias, true, sA, 160, lane, wave); __syncthreads();
  copy_w(nW3T, sW, 128 * 128, tid); copy_bias(nb3, sBias, 128, tid); __syncthreads();
  mlp_layer<4>(sA, 160, sW, sBias, false, sB, 128, lane, wave); __syncthreads();
  copy_w(dW0T, sW, 128 * 128, tid); copy_bias(db0, sBias, 128, tid); __syncthreads();
  mlp_layer<4>(sB, 128, sW, sBias, true, sA, 160, lane, wave); __syncthreads();
  copy_w(dW2T, sW, 128 * 128, tid); copy_bias(db2, sBias, 128, tid); __syncthreads();
  mlp_layer<4>(sA, 160, sW, sBias, true, sB, 128, lane, wave); __syncthreads();
  copy_w(dW3T, sW, 16 * 128, tid);  copy_bias(db3, sBias, 4, tid);   __syncthreads();

  // Final layer: [128x128] x [128x16] (only 4 real output cols), store to d_out
  {
    v16bf af[4];
    const bf16* rowp = sB + (wave * 16 + (lane & 15)) * 128;
#pragma unroll
    for (int kc = 0; kc < 4; ++kc) af[kc] = load_afrag(rowp, kc, lane);
    int col0 = lane & 15;
    int rofs = wave * 16 + ((lane & 16) ? 8 : 0);
    v8f acc = {};
#pragma unroll
    for (int kc = 0; kc < 4; ++kc) {
      v16bf b = load_bfrag(sW, 128, col0, kc, lane);
      acc = __builtin_amdgcn_wmma_f32_16x16x32_bf16(false, af[kc], false, b,
                                                    (short)0, acc, false, false);
    }
    if (col0 < 4) {
      float bs = sBias[col0];
#pragma unroll
      for (int v = 0; v < 8; ++v) {
        long g = g0 + rofs + v;
        if (g < N) out[g * 4 + col0] = acc[v] + bs;
      }
    }
  }
}

// ---------------- Prep kernels ----------------
// mode 0: wT[n][k] = W[k][n]           (zero-padded)
// mode 1: edge-W0 with feature permutation [x_src | x_dst | posdiff]:
//         col k<16 -> original row k+3; 16<=k<19 -> row k-16; else 0
__global__ void prep_w_kernel(const float* __restrict__ W, bf16* __restrict__ wT,
                              int K, int Nr, int KPAD, int total, int mode) {
  int i = blockIdx.x * 256 + threadIdx.x;
  if (i >= total) return;
  int n = i / KPAD, k = i % KPAD;
  int kk = k;
  if (mode == 1) kk = (k < 16) ? (k + 3) : ((k < 19) ? (k - 16) : 0x7fff);
  float v = (kk < K && n < Nr) ? W[kk * Nr + n] : 0.f;
  wT[i] = (bf16)v;
}
__global__ void prep_x_kernel(const float* __restrict__ x, bf16* __restrict__ xb, int n) {
  int i = blockIdx.x * 256 + threadIdx.x;
  if (i < n) xb[i] = (bf16)x[i];
}
__global__ void zero_kernel(float* __restrict__ p, long n4) {
  long i = ((long)blockIdx.x * 256 + threadIdx.x) * 4;
  if (i < n4) { float4 z = {0.f, 0.f, 0.f, 0.f}; *(float4*)(p + i) = z; }
}

extern "C" void kernel_launch(void* const* d_in, const int* in_sizes, int n_in,
                              void* d_out, int out_size, void* d_ws, size_t ws_size,
                              hipStream_t stream) {
  (void)n_in; (void)out_size; (void)ws_size;
  const float* x   = (const float*)d_in[0];
  const int*   ei  = (const int*)d_in[1];
  const float* pos = (const float*)d_in[2];
  const float* eW0 = (const float*)d_in[3];  const float* eb0 = (const float*)d_in[4];
  const float* eW2 = (const float*)d_in[5];  const float* eb2 = (const float*)d_in[6];
  const float* eW3 = (const float*)d_in[7];  const float* eb3 = (const float*)d_in[8];
  const float* nW0 = (const float*)d_in[9];  const float* nb0 = (const float*)d_in[10];
  const float* nW2 = (const float*)d_in[11]; const float* nb2 = (const float*)d_in[12];
  const float* nW3 = (const float*)d_in[13]; const float* nb3 = (const float*)d_in[14];
  const float* dW0 = (const float*)d_in[15]; const float* db0 = (const float*)d_in[16];
  const float* dW2 = (const float*)d_in[17]; const float* db2 = (const float*)d_in[18];
  const float* dW3 = (const float*)d_in[19]; const float* db3 = (const float*)d_in[20];

  int N = in_sizes[0] / 8;
  int E = in_sizes[1] / 2;
  float* out = (float*)d_out;

  char* ws = (char*)d_ws;
  size_t o = 0;
  float* es = (float*)(ws + o); o += (size_t)N * 128 * sizeof(float);
  bf16*  xb = (bf16*)(ws + o);  o += (size_t)N * 8 * sizeof(bf16);
  auto alloc_bf = [&](size_t elems) { bf16* p = (bf16*)(ws + o); o += elems * 2; return p; };
  bf16* eW0T = alloc_bf(128 * 32);
  bf16* eW2T = alloc_bf(128 * 128);
  bf16* eW3T = alloc_bf(128 * 128);
  bf16* nW0T = alloc_bf(128 * 160);
  bf16* nW2T = alloc_bf(128 * 128);
  bf16* nW3T = alloc_bf(128 * 128);
  bf16* dW0T = alloc_bf(128 * 128);
  bf16* dW2T = alloc_bf(128 * 128);
  bf16* dW3T = alloc_bf(16 * 128);

  auto prep = [&](const float* W, bf16* T, int K, int Nr, int KPAD, int NPAD, int mode) {
    int total = NPAD * KPAD;
    prep_w_kernel<<<(total + 255) / 256, 256, 0, stream>>>(W, T, K, Nr, KPAD, total, mode);
  };
  prep(eW0, eW0T, 19, 128, 32, 128, 1);   // permuted feature order
  prep(eW2, eW2T, 128, 128, 128, 128, 0);
  prep(eW3, eW3T, 128, 128, 128, 128, 0);
  prep(nW0, nW0T, 136, 128, 160, 128, 0);
  prep(nW2, nW2T, 128, 128, 128, 128, 0);
  prep(nW3, nW3T, 128, 128, 128, 128, 0);
  prep(dW0, dW0T, 128, 128, 128, 128, 0);
  prep(dW2, dW2T, 128, 128, 128, 128, 0);
  prep(dW3, dW3T, 128, 4, 128, 16, 0);

  prep_x_kernel<<<(N * 8 + 255) / 256, 256, 0, stream>>>(x, xb, N * 8);
  long esn = (long)N * 128;
  zero_kernel<<<(int)((esn / 4 + 255) / 256), 256, 0, stream>>>(es, esn);

  int ntiles = (E + 127) / 128;
  int eblocks = ntiles < 2048 ? ntiles : 2048;
  edge_kernel<<<eblocks, 256, 0, stream>>>(xb, pos, ei, E, ntiles,
                                           eW0T, eb0, eW2T, eb2, eW3T, eb3, es);
  node_kernel<<<(N + 127) / 128, 256, 0, stream>>>(xb, es, N,
                                                   nW0T, nb0, nW2T, nb2, nW3T, nb3,
                                                   dW0T, db0, dW2T, db2, dW3T, db3, out);
}
